// EncoderLayer_67224828117019
// MI455X (gfx1250) — compile-verified, matrix operand order checked by reading, not run on hardware
//
#include <hip/hip_runtime.h>
#include <hip/hip_bf16.h>

// ---------------------------------------------------------------------------
// CDNA5 (gfx1250) transformer encoder layer.
//  - All GEMMs: v_wmma_f32_16x16x32_f16 (f16 operands, f32 accumulate)
//  - Global->LDS tile staging via global_load_async_to_lds_b128 (ASYNCcnt),
//    double-buffered, shared by all 8 waves of a workgroup.
//  - Flash attention (online softmax), transposed so softmax is per-lane and
//    P^T C-fragments feed WMMA's B operand directly (no shuffles/LDS).
// ---------------------------------------------------------------------------

typedef __attribute__((ext_vector_type(16))) _Float16 v16h;
typedef __attribute__((ext_vector_type(8)))  _Float16 v8h;
typedef __attribute__((ext_vector_type(8)))  float    v8f;

union Frag16 { v16h v; v8h h[2]; _Float16 e[16]; };
union FragC  { v8f  v; float e[8]; };

__device__ __forceinline__ v8f zero8() {
  v8f z = {0.f,0.f,0.f,0.f,0.f,0.f,0.f,0.f};
  return z;
}

__device__ __forceinline__ v8f wmma16(v16h a, v16h b, v8f c) {
  // (neg_a, A, neg_b, B, c_mod, C, reuse_a, reuse_b)
  return __builtin_amdgcn_wmma_f32_16x16x32_f16(false, a, false, b, (short)0, c,
                                                false, false);
}

// --- CDNA5 async global->LDS copy (16B per lane), tracked by ASYNCcnt ------
// VDST = per-lane LDS byte offset (allocation-relative; low 32 bits of a
// generic __shared__ pointer are exactly the AS(3) offset on amdgcn).
__device__ __forceinline__ void async_copy16(const _Float16* lds_dst,
                                             const _Float16* gsrc) {
  unsigned int loff = (unsigned int)(unsigned long long)lds_dst;
  unsigned long long gaddr = (unsigned long long)gsrc;
  asm volatile("global_load_async_to_lds_b128 %0, %1, off"
               :: "v"(loff), "v"(gaddr) : "memory");
}

__device__ __forceinline__ void wait_async0() {
#if defined(__has_builtin) && __has_builtin(__builtin_amdgcn_s_wait_asynccnt)
  __builtin_amdgcn_s_wait_asynccnt(0);
#else
  asm volatile("s_wait_asynccnt 0x0" ::: "memory");
#endif
}

// A/B fragment (16x32 / 32x16 f16) from row-major GLOBAL memory.
// lane&15 -> row; elements: K = (e>=8 ? 16:0) + (lane>>4)*8 + (e&7)
__device__ __forceinline__ v16h load_frag(const _Float16* __restrict__ src,
                                          int ld, int r0, int k0) {
  const int lane = threadIdx.x & 31;
  const _Float16* p = src + (size_t)(r0 + (lane & 15)) * ld
                          + k0 + ((lane >> 4) << 3);
  Frag16 f;
  f.h[0] = *(const v8h*)(p);
  f.h[1] = *(const v8h*)(p + 16);
  return f.v;
}

// Same fragment shape from an LDS tile (row stride in halves, padded).
__device__ __forceinline__ v16h lds_frag(const _Float16* base, int stride) {
  const int lane = threadIdx.x & 31;
  const _Float16* q = base + (lane & 15) * stride + ((lane >> 4) << 3);
  Frag16 f;
  f.h[0] = *(const v8h*)(q);
  f.h[1] = *(const v8h*)(q + 16);
  return f.v;
}

// ---------------------------------------------------------------------------
__global__ void cvt_f32_f16(const float* __restrict__ in,
                            _Float16* __restrict__ out, int n) {
  int i = blockIdx.x * blockDim.x + threadIdx.x;
  if (i < n) out[i] = (_Float16)in[i];
}

// ---------------------------------------------------------------------------
// C[M][N] = A[M][K] * B[N][K]^T (+bias[n]) (+relu)
// mode 0: f32 out[M][N] ; mode 1: f16 out[M][N] ; mode 2: f16 out[N][M]
// Block 256 thr = 8 waves; block tile 128M x 64N; wave w -> 16-row M subtile.
// K stepped by 32 through double-buffered LDS tiles filled asynchronously.
// ---------------------------------------------------------------------------
#define ASTR 40  // LDS row stride (halves) for 32-half rows (+8 pad)
#define KSTR 72  // LDS row stride (halves) for 64-half rows (+8 pad)

__global__ __launch_bounds__(256) void gemm_awbT(
    const _Float16* __restrict__ A, const _Float16* __restrict__ B,
    const float* __restrict__ bias, void* __restrict__ out,
    int M, int N, int K, int mode, int relu) {
  __shared__ _Float16 As[2][128 * ASTR];
  __shared__ _Float16 Bs[2][64 * ASTR];

  const int tid = threadIdx.x;
  const int wave = tid >> 5;
  const int lane = tid & 31;
  const int m_blk = blockIdx.x * 128;
  const int n0 = blockIdx.y * 64;

  // staging map: 16B (8 halves) per lane per instruction
  const int srow = tid >> 2;        // 0..63
  const int schk = (tid & 3) * 8;   // half offset within 32-half row

  auto stage = [&](int buf, int k0) {
    async_copy16(&As[buf][srow * ASTR + schk],
                 A + (size_t)(m_blk + srow) * K + k0 + schk);
    async_copy16(&As[buf][(64 + srow) * ASTR + schk],
                 A + (size_t)(m_blk + 64 + srow) * K + k0 + schk);
    async_copy16(&Bs[buf][srow * ASTR + schk],
                 B + (size_t)(n0 + srow) * K + k0 + schk);
  };

  v8f acc[4];
#pragma unroll
  for (int j = 0; j < 4; ++j) acc[j] = zero8();

  stage(0, 0);
  wait_async0();
  __syncthreads();

  int buf = 0;
  for (int k0 = 0; k0 < K; k0 += 32) {
    if (k0 + 32 < K) stage(buf ^ 1, k0 + 32);  // fill next buffer in flight
    v16h a = lds_frag(&As[buf][(wave * 16) * ASTR], ASTR);
#pragma unroll
    for (int j = 0; j < 4; ++j) {
      v16h b = lds_frag(&Bs[buf][(j * 16) * ASTR], ASTR);
      acc[j] = wmma16(a, b, acc[j]);
    }
    wait_async0();
    __syncthreads();
    buf ^= 1;
  }

  const int nl = lane & 15;
  const int mh = (lane >> 4) << 3;
#pragma unroll
  for (int j = 0; j < 4; ++j) {
    const int n = n0 + j * 16 + nl;
    const float bv = bias ? bias[n] : 0.0f;
    FragC c; c.v = acc[j];
    if (mode == 2) {
      v8h st;
#pragma unroll
      for (int r = 0; r < 8; ++r) {
        float v = c.e[r] + bv;
        if (relu) v = fmaxf(v, 0.0f);
        st[r] = (_Float16)v;
      }
      *(v8h*)((_Float16*)out + (size_t)n * M + (m_blk + wave * 16 + mh)) = st;
    } else if (mode == 1) {
#pragma unroll
      for (int r = 0; r < 8; ++r) {
        float v = c.e[r] + bv;
        if (relu) v = fmaxf(v, 0.0f);
        ((_Float16*)out)[(size_t)(m_blk + wave * 16 + mh + r) * N + n] =
            (_Float16)v;
      }
    } else {
#pragma unroll
      for (int r = 0; r < 8; ++r) {
        float v = c.e[r] + bv;
        if (relu) v = fmaxf(v, 0.0f);
        ((float*)out)[(size_t)(m_blk + wave * 16 + mh + r) * N + n] = v;
      }
    }
  }
}

// ---------------------------------------------------------------------------
// Flash attention, transposed. One 16-query tile per wave; 8 waves/block share
// the K / V^T tiles of their head through double-buffered async LDS staging.
//   S^T   = K_mat * Q^T   (q in lanes -> per-lane softmax, one shfl_xor(16))
//   ctx^T = V^T * P^T     (P^T C-fragments are bit-identical to B operands)
// qh,kh: [S][512] f16 ; vT: [512][S] f16 ; ctx: [S][512] f16.
// ---------------------------------------------------------------------------
__global__ __launch_bounds__(256) void flash_attn(
    const _Float16* __restrict__ qh, const _Float16* __restrict__ kh,
    const _Float16* __restrict__ vT, _Float16* __restrict__ ctx, int S) {
  __shared__ _Float16 Ks[2][32 * KSTR];  // 32 keys x 64 depth (+pad)
  __shared__ _Float16 Vs[2][64 * ASTR];  // 64 depth x 32 keys (+pad)

  const int tid = threadIdx.x;
  const int wave = tid >> 5;
  const int lane = tid & 31;
  const int head = blockIdx.y;  // 0..7
  const int q0 = (blockIdx.x * 8 + wave) * 16;
  const int LD = 512;

  const _Float16* qbase = qh + head * 64;
  const _Float16* kbase = kh + head * 64;
  const _Float16* vbase = vT + (size_t)(head * 64) * S;

  // staging maps (16B per lane per instruction)
  const int krow = tid >> 3;             // 0..31 key rows (8 chunks/row)
  const int kchk = (tid & 7) * 8;        // within 64-half row
  const int vrow = tid >> 2;             // 0..63 depth rows (4 chunks/row)
  const int vchk = (tid & 3) * 8;        // within 32-half row

  auto stage = [&](int buf, int key0) {
    async_copy16(&Ks[buf][krow * KSTR + kchk],
                 kbase + (size_t)(key0 + krow) * LD + kchk);
    async_copy16(&Vs[buf][vrow * ASTR + vchk],
                 vbase + (size_t)vrow * S + key0 + vchk);
  };

  // Q^T as B operand: lane = q column, elements = depth
  const v16h bq0 = load_frag(qbase, LD, q0, 0);   // depth 0..31
  const v16h bq1 = load_frag(qbase, LD, q0, 32);  // depth 32..63

  v8f acc[4];
#pragma unroll
  for (int j = 0; j < 4; ++j) acc[j] = zero8();
  float m_run = -1e30f;
  float l_run = 0.0f;
  const float scale = 0.125f;  // 1/sqrt(64)

  stage(0, 0);
  wait_async0();
  __syncthreads();

  int buf = 0;
  for (int key0 = 0; key0 < S; key0 += 32) {
    if (key0 + 32 < S) stage(buf ^ 1, key0 + 32);

    // logits^T for 32 keys: two 16k x 16q fragments out of LDS
    FragC s0, s1;
    {
      v8f t = zero8();
      t = wmma16(lds_frag(&Ks[buf][0], KSTR), bq0, t);
      t = wmma16(lds_frag(&Ks[buf][32], KSTR), bq1, t);
      s0.v = t;
      v8f u = zero8();
      u = wmma16(lds_frag(&Ks[buf][16 * KSTR], KSTR), bq0, u);
      u = wmma16(lds_frag(&Ks[buf][16 * KSTR + 32], KSTR), bq1, u);
      s1.v = u;
    }

    // per-query (per-lane) online softmax
    float mt = -1e30f;
#pragma unroll
    for (int r = 0; r < 8; ++r) {
      s0.e[r] *= scale;
      s1.e[r] *= scale;
      mt = fmaxf(mt, fmaxf(s0.e[r], s1.e[r]));
    }
    mt = fmaxf(mt, __shfl_xor(mt, 16));  // lanes L, L+16 share a query
    const float m_new = fmaxf(m_run, mt);
    const float corr = __expf(m_run - m_new);

    float lt = 0.0f;
    Frag16 pb;  // P^T, directly in WMMA B-operand layout
#pragma unroll
    for (int r = 0; r < 8; ++r) {
      float p0 = __expf(s0.e[r] - m_new);
      float p1 = __expf(s1.e[r] - m_new);
      lt += p0 + p1;
      pb.e[r]     = (_Float16)p0;  // keys key0    + 8*(lane>>4) + r
      pb.e[r + 8] = (_Float16)p1;  // keys key0+16 + 8*(lane>>4) + r
    }
    lt += __shfl_xor(lt, 16);
    l_run = l_run * corr + lt;
    m_run = m_new;

#pragma unroll
    for (int j = 0; j < 4; ++j) {
      FragC c; c.v = acc[j];
#pragma unroll
      for (int r = 0; r < 8; ++r) c.e[r] *= corr;
      v16h av = lds_frag(&Vs[buf][(j * 16) * ASTR], ASTR);  // V^T rows = depth
      acc[j] = wmma16(av, pb.v, c.v);
    }

    wait_async0();
    __syncthreads();
    buf ^= 1;
  }

  // ctx^T fragments: lane = q, elements = contiguous depth -> v8h stores
  const float inv = 1.0f / l_run;
  const int ql = lane & 15;
  const int dh = (lane >> 4) << 3;
#pragma unroll
  for (int j = 0; j < 4; ++j) {
    FragC c; c.v = acc[j];
    v8h st;
#pragma unroll
    for (int r = 0; r < 8; ++r) st[r] = (_Float16)(c.e[r] * inv);
    *(v8h*)(ctx + (size_t)(q0 + ql) * LD + head * 64 + j * 16 + dh) = st;
  }
}

// ---------------------------------------------------------------------------
// out = LayerNorm(resid + y) * g + b ; optional f16 copy. One row per block.
// ---------------------------------------------------------------------------
__global__ __launch_bounds__(256) void add_layernorm(
    const float* __restrict__ resid, const float* __restrict__ y,
    const float* __restrict__ g, const float* __restrict__ b,
    float* __restrict__ out32, _Float16* __restrict__ out16, int N) {
  __shared__ float red[16];
  const int row = blockIdx.x;
  const int i0 = threadIdx.x;
  const int i1 = threadIdx.x + 256;
  const size_t base = (size_t)row * N;

  float v0 = resid[base + i0] + y[base + i0];
  float v1 = resid[base + i1] + y[base + i1];
  float s = v0 + v1;
  float s2 = v0 * v0 + v1 * v1;
#pragma unroll
  for (int off = 16; off > 0; off >>= 1) {
    s  += __shfl_xor(s, off);
    s2 += __shfl_xor(s2, off);
  }
  const int wave = threadIdx.x >> 5;
  if ((threadIdx.x & 31) == 0) { red[wave] = s; red[wave + 8] = s2; }
  __syncthreads();
  if (threadIdx.x == 0) {
    float ts = 0.f, ts2 = 0.f;
    for (int i = 0; i < 8; ++i) { ts += red[i]; ts2 += red[i + 8]; }
    red[0] = ts; red[8] = ts2;
  }
  __syncthreads();
  const float mean = red[0] / (float)N;
  const float var  = red[8] / (float)N - mean * mean;
  const float rstd = rsqrtf(var + 1e-6f);

  float o0 = (v0 - mean) * rstd * g[i0] + b[i0];
  float o1 = (v1 - mean) * rstd * g[i1] + b[i1];
  out32[base + i0] = o0;
  out32[base + i1] = o1;
  if (out16) {
    out16[base + i0] = (_Float16)o0;
    out16[base + i1] = (_Float16)o1;
  }
}

// ---------------------------------------------------------------------------
extern "C" void kernel_launch(void* const* d_in, const int* in_sizes, int n_in,
                              void* d_out, int out_size, void* d_ws,
                              size_t ws_size, hipStream_t stream) {
  (void)in_sizes; (void)n_in; (void)out_size; (void)ws_size;
  const int S = 4096, DM = 512, DFF = 2048;

  const float* x       = (const float*)d_in[0];
  const float* wq_w    = (const float*)d_in[1];
  const float* wq_b    = (const float*)d_in[2];
  const float* wk_w    = (const float*)d_in[3];
  const float* wk_b    = (const float*)d_in[4];
  const float* wv_w    = (const float*)d_in[5];
  const float* wv_b    = (const float*)d_in[6];
  const float* dense_w = (const float*)d_in[7];
  const float* dense_b = (const float*)d_in[8];
  const float* l1_w    = (const float*)d_in[9];
  const float* l1_b    = (const float*)d_in[10];
  const float* l2_w    = (const float*)d_in[11];
  const float* l2_b    = (const float*)d_in[12];
  const float* ln1_g   = (const float*)d_in[13];
  const float* ln1_b   = (const float*)d_in[14];
  const float* ln2_g   = (const float*)d_in[15];
  const float* ln2_b   = (const float*)d_in[16];
  float* out = (float*)d_out;

  char* base = (char*)d_ws;
  size_t off = 0;
  auto a16 = [&](size_t elems) -> _Float16* {
    _Float16* p = (_Float16*)(base + off);
    off = (off + elems * sizeof(_Float16) + 255) & ~(size_t)255;
    return p;
  };
  auto a32 = [&](size_t elems) -> float* {
    float* p = (float*)(base + off);
    off = (off + elems * sizeof(float) + 255) & ~(size_t)255;
    return p;
  };

  _Float16* xh     = a16((size_t)S * DM);
  _Float16* wqh    = a16((size_t)DM * DM);
  _Float16* wkh    = a16((size_t)DM * DM);
  _Float16* wvh    = a16((size_t)DM * DM);
  _Float16* denseh = a16((size_t)DM * DM);
  _Float16* l1h    = a16((size_t)DFF * DM);
  _Float16* l2h    = a16((size_t)DM * DFF);
  _Float16* qhb    = a16((size_t)S * DM);
  _Float16* khb    = a16((size_t)S * DM);
  _Float16* vTh    = a16((size_t)DM * S);
  _Float16* ctxh   = a16((size_t)S * DM);
  _Float16* out1h  = a16((size_t)S * DM);
  _Float16* h1h    = a16((size_t)S * DFF);
  float*    attn_o = a32((size_t)S * DM);  // reused for FFN2 output
  float*    out1   = a32((size_t)S * DM);

  auto cvt = [&](const float* src, _Float16* dst, int n) {
    cvt_f32_f16<<<(n + 255) / 256, 256, 0, stream>>>(src, dst, n);
  };
  cvt(x,       xh,     S * DM);
  cvt(wq_w,    wqh,    DM * DM);
  cvt(wk_w,    wkh,    DM * DM);
  cvt(wv_w,    wvh,    DM * DM);
  cvt(dense_w, denseh, DM * DM);
  cvt(l1_w,    l1h,    DFF * DM);
  cvt(l2_w,    l2h,    DM * DFF);

  const dim3 blk(256);
  // Q/K/V projections (V stored transposed)
  gemm_awbT<<<dim3(S / 128, DM / 64), blk, 0, stream>>>(
      xh, wqh, wq_b, qhb, S, DM, DM, /*mode=*/1, /*relu=*/0);
  gemm_awbT<<<dim3(S / 128, DM / 64), blk, 0, stream>>>(
      xh, wkh, wk_b, khb, S, DM, DM, 1, 0);
  gemm_awbT<<<dim3(S / 128, DM / 64), blk, 0, stream>>>(
      xh, wvh, wv_b, vTh, S, DM, DM, /*mode=*/2, 0);

  // flash attention (8 heads)
  flash_attn<<<dim3(S / 128, 8), blk, 0, stream>>>(qhb, khb, vTh, ctxh, S);

  // dense projection
  gemm_awbT<<<dim3(S / 128, DM / 64), blk, 0, stream>>>(
      ctxh, denseh, dense_b, attn_o, S, DM, DM, /*mode=*/0, 0);

  // residual + LayerNorm 1 (also emits f16 activations)
  add_layernorm<<<S, blk, 0, stream>>>(x, attn_o, ln1_g, ln1_b, out1, out1h, DM);

  // FFN
  gemm_awbT<<<dim3(S / 128, DFF / 64), blk, 0, stream>>>(
      out1h, l1h, l1_b, h1h, S, DFF, DM, /*mode=*/1, /*relu=*/1);
  gemm_awbT<<<dim3(S / 128, DM / 64), blk, 0, stream>>>(
      h1h, l2h, l2_b, attn_o, S, DM, DFF, /*mode=*/0, 0);

  // residual + LayerNorm 2 -> d_out
  add_layernorm<<<S, blk, 0, stream>>>(out1, attn_o, ln2_g, ln2_b, out, nullptr, DM);
}
